// MultiHeadAttention_91018946937263
// MI455X (gfx1250) — matrix operand order repaired and checked by reading, not compiled, not run on hardware
//
#include <hip/hip_runtime.h>
#include <hip/hip_bf16.h>

// Problem constants (match reference)
#define B_      32
#define S_      4096
#define H_      1024
#define NH_     16
#define DK_     64
#define NSPLIT  16
#define SSPLIT  (S_ / NSPLIT)   // 256 positions per split
#define SC      64              // chunk of sequence rows per iteration
#define NCHUNK  (SSPLIT / SC)   // 4

typedef __attribute__((ext_vector_type(16))) __bf16 v16bf;
typedef __attribute__((ext_vector_type(8)))  float  v8f;

union AB { v16bf v; uint4 q[2]; };

// Native bf16 convert (RTNE) -> hardware v_cvt on gfx1250
__device__ __forceinline__ unsigned short f2bf(float f) {
  union { __bf16 h; unsigned short u; } v;
  v.h = (__bf16)f;
  return v.u;
}

__device__ __forceinline__ v8f wmma_bf16(const AB& a, const AB& b, v8f c) {
  return __builtin_amdgcn_wmma_f32_16x16x32_bf16(
      /*neg_a=*/false, a.v, /*neg_b=*/false, b.v,
      /*c_mod=*/(short)0, c, /*reuse_a=*/false, /*reuse_b=*/false);
}

// ---------------------------------------------------------------------------
// q[b,n] = x[b,S-1,:]·Wq[n,:] + bq[n]   (tiny: 32x1024 outputs)
// ---------------------------------------------------------------------------
__global__ __launch_bounds__(256) void prepq_kernel(
    const float* __restrict__ x, const float* __restrict__ Wq,
    const float* __restrict__ bq, float* __restrict__ qf) {
  int idx = blockIdx.x * 256 + threadIdx.x;       // < 32*1024
  int n = idx & (H_ - 1), b = idx >> 10;
  const float* xr = x + ((size_t)b * S_ + (S_ - 1)) * H_;
  const float* wr = Wq + (size_t)n * H_;
  float a = 0.f;
  for (int c = 0; c < H_; ++c) a += xr[c] * wr[c];
  qf[idx] = a + bq[n];
}

// ---------------------------------------------------------------------------
// wqk[b,h,c] = scale * Σ_d q[b,h*64+d]·Wk[h*64+d,c]  (bf16)
// qbk[b,h]   = scale * q_h·bk_h
// ---------------------------------------------------------------------------
__global__ __launch_bounds__(256) void prepwqk_kernel(
    const float* __restrict__ qf, const float* __restrict__ Wk,
    const float* __restrict__ bk, unsigned short* __restrict__ wqkb,
    float* __restrict__ qbk) {
  int idx = blockIdx.x * 256 + threadIdx.x;       // < 32*16*1024
  int c = idx & (H_ - 1), h = (idx >> 10) & (NH_ - 1), b = idx >> 14;
  const float scale = 0.125f;                     // dk^-0.5 = 1/8
  const float* qh = qf + b * H_ + h * DK_;
  float a = 0.f;
  for (int d = 0; d < DK_; ++d) a += qh[d] * Wk[(size_t)(h * DK_ + d) * H_ + c];
  wqkb[idx] = f2bf(a * scale);
  if (c == 0) {
    float s = 0.f;
    for (int d = 0; d < DK_; ++d) s += qh[d] * bk[h * DK_ + d];
    qbk[b * NH_ + h] = s * scale;
  }
}

// ---------------------------------------------------------------------------
// Convert Wv and Wo to bf16 (row-major, K=H contiguous -> direct WMMA B reads)
// ---------------------------------------------------------------------------
__global__ __launch_bounds__(256) void conv_kernel(
    const float* __restrict__ Wv, const float* __restrict__ Wo,
    unsigned short* __restrict__ Wvb, unsigned short* __restrict__ Wob) {
  int idx = blockIdx.x * 256 + threadIdx.x;       // < 2*H*H
  if (idx < H_ * H_) Wvb[idx] = f2bf(Wv[idx]);
  else               Wob[idx - H_ * H_] = f2bf(Wo[idx - H_ * H_]);
}

// ---------------------------------------------------------------------------
// Fused flash-decode over one S-split of one batch.
//  grid = (NSPLIT, B), 256 threads (8 waves).
//  Per chunk: x(64xH) -> LDS (bf16, row- AND col-major),
//    scores^T(16hx64s) = wqk(16xH) @ x^T   [WMMA, split-K, ds_add_f32 reduce]
//    online softmax per head
//    Y(16hx1024c) += P(16x64) @ x(64x1024) [WMMA, accumulators in VGPRs]
// ---------------------------------------------------------------------------
__global__ __launch_bounds__(256) void flash_kernel(
    const float* __restrict__ x, const unsigned short* __restrict__ wqk,
    const float* __restrict__ qbk, float* __restrict__ Yp,
    float* __restrict__ mp, float* __restrict__ lp) {
  const int b = blockIdx.y, split = blockIdx.x;
  const int tid = threadIdx.x;
  const int wave = tid >> 5, lane = tid & 31;
  const int hi = lane >> 4, ln = lane & 15;

  // ~318 KB of the 320 KB WGP LDS
  __shared__ unsigned short xRM[SC][H_ + 8];      // x chunk, row-major (pad 16B)
  __shared__ unsigned short xCM[H_][SC + 8];      // x chunk, col-major (pad 16B)
  __shared__ unsigned short wqkL[NH_][H_];        // per-batch WQK panel
  __shared__ float          sc[NH_][SC];          // scores^T
  __shared__ unsigned short P[NH_][SC];           // softmax weights (bf16)
  __shared__ float mS[NH_], lS[NH_], aS[NH_], qbS[NH_];

  for (int i = tid; i < NH_ * H_; i += 256)
    wqkL[0][i] = wqk[(size_t)b * NH_ * H_ + i];
  if (tid < NH_) {
    mS[tid] = -1e30f; lS[tid] = 0.0f;
    qbS[tid] = qbk[b * NH_ + tid];
  }

  const v8f vzero = {0.f, 0.f, 0.f, 0.f, 0.f, 0.f, 0.f, 0.f};
  v8f acc[8];                                     // Y: 8 c-tiles per wave
  for (int t = 0; t < 8; ++t) acc[t] = vzero;

  const float* xb = x + ((size_t)b * S_ + (size_t)split * SSPLIT) * H_;

  for (int ch = 0; ch < NCHUNK; ++ch) {
    __syncthreads();                              // LDS safe to overwrite
    // ---- stage x chunk: f32 -> bf16, both layouts ----
    const float* xc = xb + (size_t)ch * SC * H_;
    for (int i = tid; i < SC * (H_ / 4); i += 256) {
      int s = i >> 8;                             // H/4 = 256 float4 per row
      int c4 = (i & 255) << 2;
      float4 f = *(const float4*)(xc + (size_t)s * H_ + c4);
      if (ch + 1 < NCHUNK)
        __builtin_prefetch(xc + (size_t)SC * H_ + (size_t)s * H_ + c4, 0, 1);
      unsigned short u0 = f2bf(f.x), u1 = f2bf(f.y), u2 = f2bf(f.z), u3 = f2bf(f.w);
      *(unsigned int*)&xRM[s][c4]     = (unsigned)u0 | ((unsigned)u1 << 16);
      *(unsigned int*)&xRM[s][c4 + 2] = (unsigned)u2 | ((unsigned)u3 << 16);
      xCM[c4 + 0][s] = u0; xCM[c4 + 1][s] = u1;
      xCM[c4 + 2][s] = u2; xCM[c4 + 3][s] = u3;
    }
    // init scores with q·bk bias (split-K waves ds_add into this)
    for (int j = 0; j < (NH_ * SC) / 256; ++j) {
      int idx = j * 256 + tid;
      sc[idx >> 6][idx & (SC - 1)] = qbS[idx >> 6];
    }
    __syncthreads();

    // ---- GEMM1: scores^T = WQK^T @ x^T ; M=h(16), N=s, K=c ----
    {
      const int stile = (wave & 3) * 16;          // 4 s-tiles
      const int kbase = (wave >> 2) * (H_ / 2);   // split-K halves
      v8f c1 = vzero;
      const unsigned short* ar = &wqkL[ln][0];            // A: row h=ln
      const unsigned short* br = &xRM[stile + ln][0];     // B: col s, K=c contig
      #pragma unroll 4
      for (int k0 = kbase; k0 < kbase + H_ / 2; k0 += 32) {
        AB a, bm;
        a.q[0]  = *(const uint4*)(ar + k0 + hi * 8);
        a.q[1]  = *(const uint4*)(ar + k0 + hi * 8 + 16);
        bm.q[0] = *(const uint4*)(br + k0 + hi * 8);
        bm.q[1] = *(const uint4*)(br + k0 + hi * 8 + 16);
        c1 = wmma_bf16(a, bm, c1);
      }
      const int s2 = stile + ln;
      #pragma unroll
      for (int r = 0; r < 8; ++r)
        atomicAdd(&sc[r + 8 * hi][s2], c1[r]);    // ds_add_f32 split-K reduce
    }
    __syncthreads();

    // ---- online softmax per head ----
    if (tid < NH_) {
      const int h = tid;
      float mold = mS[h], mloc = mold;
      for (int s2 = 0; s2 < SC; ++s2) mloc = fmaxf(mloc, sc[h][s2]);
      float alpha = __expf(mold - mloc);
      float sum = 0.f;
      for (int s2 = 0; s2 < SC; ++s2) {
        float p = __expf(sc[h][s2] - mloc);
        P[h][s2] = f2bf(p);
        sum += p;
      }
      lS[h] = lS[h] * alpha + sum;
      mS[h] = mloc;
      aS[h] = alpha;
    }
    __syncthreads();

    // ---- GEMM2: Y = alpha*Y + P @ x_chunk ; M=h(16), N=c, K=s ----
    float av[8];
    #pragma unroll
    for (int r = 0; r < 8; ++r) av[r] = aS[r + 8 * hi];
    const unsigned short* pr = &P[ln][0];         // A: row h=ln
    #pragma unroll
    for (int t = 0; t < 8; ++t) {
      v8f yy = acc[t];
      #pragma unroll
      for (int r = 0; r < 8; ++r) yy[r] *= av[r];
      const int c = (wave * 8 + t) * 16 + ln;
      const unsigned short* br = &xCM[c][0];      // B: col c, K=s contig
      #pragma unroll
      for (int k0 = 0; k0 < SC; k0 += 32) {
        AB a, bm;
        a.q[0]  = *(const uint4*)(pr + k0 + hi * 8);
        a.q[1]  = *(const uint4*)(pr + k0 + hi * 8 + 16);
        bm.q[0] = *(const uint4*)(br + k0 + hi * 8);
        bm.q[1] = *(const uint4*)(br + k0 + hi * 8 + 16);
        yy = wmma_bf16(a, bm, yy);
      }
      acc[t] = yy;
    }
  }

  // ---- write split partials ----
  float* yp = Yp + ((size_t)(b * NSPLIT + split)) * NH_ * H_;
  for (int t = 0; t < 8; ++t) {
    const int c = (wave * 8 + t) * 16 + ln;
    #pragma unroll
    for (int r = 0; r < 8; ++r)
      yp[(r + 8 * hi) * H_ + c] = acc[t][r];
  }
  if (tid < NH_) {
    mp[(b * NSPLIT + split) * NH_ + tid] = mS[tid];
    lp[(b * NSPLIT + split) * NH_ + tid] = lS[tid];
  }
}

// ---------------------------------------------------------------------------
// Combine split partials: y[b,h,c] = Σ_i Y_i e^{m_i-M} / L   (bf16 out)
// ---------------------------------------------------------------------------
__global__ __launch_bounds__(256) void combine_kernel(
    const float* __restrict__ Yp, const float* __restrict__ mp,
    const float* __restrict__ lp, unsigned short* __restrict__ ycb) {
  int idx = blockIdx.x * 256 + threadIdx.x;       // < 32*16*1024
  int c = idx & (H_ - 1);
  int h = (idx >> 10) & (NH_ - 1);
  int b = idx >> 14;
  float M = -1e30f;
  for (int i = 0; i < NSPLIT; ++i)
    M = fmaxf(M, mp[(b * NSPLIT + i) * NH_ + h]);
  float L = 0.f, y = 0.f;
  for (int i = 0; i < NSPLIT; ++i) {
    float w = __expf(mp[(b * NSPLIT + i) * NH_ + h] - M);
    L += lp[(b * NSPLIT + i) * NH_ + h] * w;
    y += Yp[((size_t)(b * NSPLIT + i) * NH_ + h) * H_ + c] * w;
  }
  ycb[idx] = f2bf(y / L);
}

// ---------------------------------------------------------------------------
// attn[b, h*64+d] = y[b,h,:]·Wv[h*64+d,:] + bv    (WMMA per head; bf16 out)
// ---------------------------------------------------------------------------
__global__ __launch_bounds__(128) void projv_kernel(
    const unsigned short* __restrict__ ycb, const unsigned short* __restrict__ Wvb,
    const float* __restrict__ bv, unsigned short* __restrict__ attnb) {
  const int h = blockIdx.x;
  const int tid = threadIdx.x, wave = tid >> 5, lane = tid & 31;
  const int hi = lane >> 4, ln = lane & 15;
  const v8f vzero = {0.f, 0.f, 0.f, 0.f, 0.f, 0.f, 0.f, 0.f};
  for (int i = 0; i < 2; ++i) {
    const int tt = wave * 2 + i;                  // 8 tiles: 2 M x 4 N
    const int Mt = tt >> 2, Nt = tt & 3;
    v8f cacc = vzero;
    const unsigned short* ar = ycb + ((size_t)(Mt * 16 + ln) * NH_ + h) * H_;
    const unsigned short* br = Wvb + (size_t)(h * DK_ + Nt * 16 + ln) * H_;
    for (int k0 = 0; k0 < H_; k0 += 32) {
      AB a, bm;
      a.q[0]  = *(const uint4*)(ar + k0 + hi * 8);
      a.q[1]  = *(const uint4*)(ar + k0 + hi * 8 + 16);
      bm.q[0] = *(const uint4*)(br + k0 + hi * 8);
      bm.q[1] = *(const uint4*)(br + k0 + hi * 8 + 16);
      cacc = wmma_bf16(a, bm, cacc);
    }
    const int d = h * DK_ + Nt * 16 + ln;
    const float bvd = bv[d];
    #pragma unroll
    for (int r = 0; r < 8; ++r) {
      const int bb = Mt * 16 + r + 8 * hi;
      attnb[bb * H_ + d] = f2bf(cacc[r] + bvd);
    }
  }
}

// ---------------------------------------------------------------------------
// out = attn @ Wo^T + bo   (M=32, N=1024, K=1024; WMMA)
// ---------------------------------------------------------------------------
__global__ __launch_bounds__(64) void projo_kernel(
    const unsigned short* __restrict__ attnb, const unsigned short* __restrict__ Wob,
    const float* __restrict__ bo, float* __restrict__ out) {
  const int Nt = blockIdx.x;                      // 0..63
  const int tid = threadIdx.x, wave = tid >> 5, lane = tid & 31;
  const int hi = lane >> 4, ln = lane & 15;
  const v8f vzero = {0.f, 0.f, 0.f, 0.f, 0.f, 0.f, 0.f, 0.f};
  v8f cacc = vzero;
  const unsigned short* ar = attnb + (size_t)(wave * 16 + ln) * H_;
  const unsigned short* br = Wob + (size_t)(Nt * 16 + ln) * H_;
  for (int k0 = 0; k0 < H_; k0 += 32) {
    AB a, bm;
    a.q[0]  = *(const uint4*)(ar + k0 + hi * 8);
    a.q[1]  = *(const uint4*)(ar + k0 + hi * 8 + 16);
    bm.q[0] = *(const uint4*)(br + k0 + hi * 8);
    bm.q[1] = *(const uint4*)(br + k0 + hi * 8 + 16);
    cacc = wmma_bf16(a, bm, cacc);
  }
  const int j = Nt * 16 + ln;
  const float boj = bo[j];
  #pragma unroll
  for (int r = 0; r < 8; ++r)
    out[(wave * 16 + r + 8 * hi) * H_ + j] = cacc[r] + boj;
}

// ---------------------------------------------------------------------------
extern "C" void kernel_launch(void* const* d_in, const int* in_sizes, int n_in,
                              void* d_out, int out_size, void* d_ws, size_t ws_size,
                              hipStream_t stream) {
  (void)in_sizes; (void)n_in; (void)out_size; (void)ws_size;
  const float* x  = (const float*)d_in[0];
  const float* Wq = (const float*)d_in[1];
  const float* bq = (const float*)d_in[2];
  const float* Wk = (const float*)d_in[3];
  const float* bk = (const float*)d_in[4];
  const float* Wv = (const float*)d_in[5];
  const float* bv = (const float*)d_in[6];
  const float* Wo = (const float*)d_in[7];
  const float* bo = (const float*)d_in[8];
  float* out = (float*)d_out;

  char* ws = (char*)d_ws;
  size_t off = 0;
  auto alloc = [&](size_t bytes) {
    off = (off + 255) & ~(size_t)255;
    char* p = ws + off; off += bytes; return p;
  };
  float*          qf    = (float*)alloc((size_t)B_ * H_ * 4);
  unsigned short* wqkb  = (unsigned short*)alloc((size_t)B_ * NH_ * H_ * 2);
  float*          qbk   = (float*)alloc((size_t)B_ * NH_ * 4);
  unsigned short* Wvb   = (unsigned short*)alloc((size_t)H_ * H_ * 2);
  unsigned short* Wob   = (unsigned short*)alloc((size_t)H_ * H_ * 2);
  float*          Yp    = (float*)alloc((size_t)B_ * NSPLIT * NH_ * H_ * 4);
  float*          mp    = (float*)alloc((size_t)B_ * NSPLIT * NH_ * 4);
  float*          lp    = (float*)alloc((size_t)B_ * NSPLIT * NH_ * 4);
  unsigned short* ycb   = (unsigned short*)alloc((size_t)B_ * NH_ * H_ * 2);
  unsigned short* attnb = (unsigned short*)alloc((size_t)B_ * H_ * 2);

  prepq_kernel  <<<(B_ * H_) / 256, 256, 0, stream>>>(x, Wq, bq, qf);
  conv_kernel   <<<(2 * H_ * H_) / 256, 256, 0, stream>>>(Wv, Wo, Wvb, Wob);
  prepwqk_kernel<<<(B_ * NH_ * H_) / 256, 256, 0, stream>>>(qf, Wk, bk, wqkb, qbk);
  flash_kernel  <<<dim3(NSPLIT, B_), 256, 0, stream>>>(x, wqkb, qbk, Yp, mp, lp);
  combine_kernel<<<(B_ * NH_ * H_) / 256, 256, 0, stream>>>(Yp, mp, lp, ycb);
  projv_kernel  <<<NH_, 128, 0, stream>>>(ycb, Wvb, bv, attnb);
  projo_kernel  <<<H_ / 16, 64, 0, stream>>>(attnb, Wob, bo, out);
}
